// MSCLoss_72679436583430
// MI455X (gfx1250) — compile-verified, matrix operand order checked by reading, not run on hardware
//
#include <hip/hip_runtime.h>
#include <hip/hip_bf16.h>

#define NSRC 8192
#define NTGT 6144
#define DIM  256
#define TOPSEL 4096   // n_tgt * 2 / 3

typedef __attribute__((ext_vector_type(16))) _Float16 v16h;
typedef __attribute__((ext_vector_type(8)))  float    v8f;

struct alignas(16) F4 { float x, y, z, w; };
struct alignas(16) I4 { int   x, y, z, w; };

union FragAB { v16h v; F4 f[2]; };
union FragC  { v8f  v; F4 f[2]; };

// ---------------------------------------------------------------------------
// Kernel 1: row L2-normalize + split fp32 -> (hi, lo) f16 pair
// ---------------------------------------------------------------------------
__global__ __launch_bounds__(256)
void normalize_split(const float* __restrict__ x, _Float16* __restrict__ hi,
                     _Float16* __restrict__ lo) {
  __shared__ float red[256];
  const int row = blockIdx.x;
  const int t = threadIdx.x;
  const float v = x[(size_t)row * DIM + t];
  red[t] = v * v;
  __syncthreads();
  for (int s = 128; s > 0; s >>= 1) {
    if (t < s) red[t] += red[t + s];
    __syncthreads();
  }
  const float norm = sqrtf(red[0]);
  const float y = v / fmaxf(norm, 1e-12f);
  const _Float16 h = (_Float16)y;
  hi[(size_t)row * DIM + t] = h;
  lo[(size_t)row * DIM + t] = (_Float16)(y - (float)h);
}

// ---------------------------------------------------------------------------
// WMMA fragment load: 16x32 f16 tile per ISA 7.12.2 A-layout
// lane L<16 : row M=L,    K = k0+0..7  (VGPR0-3) and k0+16..23 (VGPR4-7)
// lane L>=16: row M=L-16, K = k0+8..15             and k0+24..31
// Both A (source rows) and B (target rows, since sim = A * B^T) use this.
// ---------------------------------------------------------------------------
__device__ inline v16h load_frag16x32(const _Float16* __restrict__ base,
                                      int row0, int k0, int lane) {
  const int r = lane & 15;
  const int h = lane >> 4;
  const _Float16* p = base + (size_t)(row0 + r) * DIM + k0 + 8 * h;
  FragAB f;
  f.f[0] = *reinterpret_cast<const F4*>(p);        // K block 0
  f.f[1] = *reinterpret_cast<const F4*>(p + 16);   // K block +16
  return f.v;
}

// ---------------------------------------------------------------------------
// Kernel 2: sim^T[t, s] = dot(a_s, b_t) via split-f16 WMMA (fp32 accuracy).
// Block: 256 thr = 8 waves, wave grid 2(M=src) x 4(N=tgt), wave tile 32x32.
// Epilogue: stage 32x32 tiles in LDS (padded), then fully coalesced
// b128 stores of the 64x128 block tile into target-major simT[t*NSRC + s].
// ---------------------------------------------------------------------------
#define TSTRIDE 36  // padded LDS row stride (words) to avoid bank conflicts

__global__ __launch_bounds__(256)
void gemm_sim(const _Float16* __restrict__ Ahi, const _Float16* __restrict__ Alo,
              const _Float16* __restrict__ Bhi, const _Float16* __restrict__ Blo,
              float* __restrict__ simT) {
  __shared__ float stile[8][32 * TSTRIDE];  // 36 KB

  const int lane = threadIdx.x & 31;
  const int wave = threadIdx.x >> 5;
  const int wm = wave & 1;
  const int wn = wave >> 1;
  const int s_base = blockIdx.x * 64  + wm * 32;
  const int t_base = blockIdx.y * 128 + wn * 32;

  v8f c[2][2] = {};

  for (int k0 = 0; k0 < DIM; k0 += 32) {
    v16h ah[2], al[2], bh[2], bl[2];
#pragma unroll
    for (int i = 0; i < 2; ++i) {
      ah[i] = load_frag16x32(Ahi, s_base + 16 * i, k0, lane);
      al[i] = load_frag16x32(Alo, s_base + 16 * i, k0, lane);
    }
#pragma unroll
    for (int j = 0; j < 2; ++j) {
      bh[j] = load_frag16x32(Bhi, t_base + 16 * j, k0, lane);
      bl[j] = load_frag16x32(Blo, t_base + 16 * j, k0, lane);
    }
#pragma unroll
    for (int i = 0; i < 2; ++i)
#pragma unroll
      for (int j = 0; j < 2; ++j) {
        c[i][j] = __builtin_amdgcn_wmma_f32_16x16x32_f16(
            false, ah[i], false, bh[j], (short)0, c[i][j], false, false);
        c[i][j] = __builtin_amdgcn_wmma_f32_16x16x32_f16(
            false, ah[i], false, bl[j], (short)0, c[i][j], false, false);
        c[i][j] = __builtin_amdgcn_wmma_f32_16x16x32_f16(
            false, al[i], false, bh[j], (short)0, c[i][j], false, false);
      }
  }

  // C/D layout: lane n<16 -> N=n, M=vgpr; lane n>=16 -> N=n-16, M=vgpr+8.
  // Stage wave tile in LDS: local layout (lt, ls) -> stile[w][lt*TSTRIDE + ls]
  const int n = lane & 15;
  const int moff = (lane >> 4) * 8;
#pragma unroll
  for (int i = 0; i < 2; ++i)
#pragma unroll
    for (int j = 0; j < 2; ++j) {
      FragC fc; fc.v = c[i][j];
      float* p = &stile[wave][(16 * j + n) * TSTRIDE + 16 * i + moff];
      *reinterpret_cast<F4*>(p)     = fc.f[0];
      *reinterpret_cast<F4*>(p + 4) = fc.f[1];
    }
  __syncthreads();

  // Coalesced writeout: 128 t-rows x 64 s-floats = 2048 float4 chunks.
  // 16 consecutive threads cover one full contiguous 256B t-row.
  const int t0 = blockIdx.y * 128;
  const int s0 = blockIdx.x * 64;
#pragma unroll
  for (int it = 0; it < 8; ++it) {
    const int cidx = (int)threadIdx.x + 256 * it;  // 0..2047
    const int t_local = cidx >> 4;                 // 0..127
    const int s_chunk = cidx & 15;                 // 0..15 (x4 floats)
    const int w  = (s_chunk >> 3) + 2 * (t_local >> 5);
    const int lt = t_local & 31;
    const int ls = (s_chunk & 7) * 4;
    const F4 val = *reinterpret_cast<const F4*>(&stile[w][lt * TSTRIDE + ls]);
    *reinterpret_cast<F4*>(simT + (size_t)(t0 + t_local) * NSRC + s0 + s_chunk * 4) = val;
  }
}

// ---------------------------------------------------------------------------
// Top-K helpers (value desc, tie -> smaller source index, matching stable
// argsort(-sim))
// ---------------------------------------------------------------------------
template <int K>
__device__ inline void insert_top(float (&v)[K], int (&ix)[K], float x, int s) {
  const float lv = v[K - 1]; const int li = ix[K - 1];
  if (!((x > lv) || (x == lv && s < li))) return;
  v[K - 1] = x; ix[K - 1] = s;
#pragma unroll
  for (int k = K - 1; k > 0; --k) {
    const bool sw = (v[k] > v[k - 1]) || (v[k] == v[k - 1] && ix[k] < ix[k - 1]);
    if (sw) {
      float tv = v[k]; v[k] = v[k - 1]; v[k - 1] = tv;
      int ti = ix[k]; ix[k] = ix[k - 1]; ix[k - 1] = ti;
    }
  }
}

template <int K>
__device__ inline void tree_merge(float* sv, int* si, int tid) {
  for (int stride = 128; stride > 0; stride >>= 1) {
    __syncthreads();
    if (tid < stride) {
      float mv[2 * K]; int mi[2 * K];
#pragma unroll
      for (int k = 0; k < K; ++k) {
        mv[k] = sv[tid * K + k];                 mi[k] = si[tid * K + k];
        mv[K + k] = sv[(tid + stride) * K + k];  mi[K + k] = si[(tid + stride) * K + k];
      }
#pragma unroll
      for (int k = 0; k < K; ++k) {
        int best = k;
#pragma unroll
        for (int j = k + 1; j < 2 * K; ++j)
          if (mv[j] > mv[best] || (mv[j] == mv[best] && mi[j] < mi[best])) best = j;
        float tv = mv[k]; mv[k] = mv[best]; mv[best] = tv;
        int ti = mi[k]; mi[k] = mi[best]; mi[best] = ti;
        sv[tid * K + k] = mv[k]; si[tid * K + k] = mi[k];
      }
    }
  }
  __syncthreads();
}

// ---------------------------------------------------------------------------
// Kernel 3: per target column -- top-5 -> majority label (min-label tiebreak),
// column max, then nln = sum(top-4 sims among positives),
// nun = sum(top-4 among negatives), score = nln/nun. b128 streaming loads.
// ---------------------------------------------------------------------------
__global__ __launch_bounds__(256)
void column_stats(const float* __restrict__ simT, const int* __restrict__ labels,
                  int* __restrict__ assigned, float* __restrict__ colmax,
                  float* __restrict__ scorev) {
  __shared__ float sv[256 * 5];
  __shared__ int   si[256 * 5];
  __shared__ int   sh_asg;
  __shared__ float sh_nln;
  const int t = blockIdx.x;
  const int tid = threadIdx.x;
  const F4* __restrict__ col4 = reinterpret_cast<const F4*>(simT + (size_t)t * NSRC);
  const I4* __restrict__ lab4 = reinterpret_cast<const I4*>(labels);

  // ---- phase 1: top-5 over the column ----
  float v5[5]; int i5[5];
#pragma unroll
  for (int k = 0; k < 5; ++k) { v5[k] = -3.4e38f; i5[k] = 0x7fffffff; }
  for (int q = tid; q < NSRC / 4; q += 256) {
    const F4 x = col4[q];
    const int s = q * 4;
    insert_top<5>(v5, i5, x.x, s);
    insert_top<5>(v5, i5, x.y, s + 1);
    insert_top<5>(v5, i5, x.z, s + 2);
    insert_top<5>(v5, i5, x.w, s + 3);
  }
#pragma unroll
  for (int k = 0; k < 5; ++k) { sv[tid * 5 + k] = v5[k]; si[tid * 5 + k] = i5[k]; }
  tree_merge<5>(sv, si, tid);

  if (tid == 0) {
    int lab[5];
#pragma unroll
    for (int k = 0; k < 5; ++k) lab[k] = labels[si[k]];
    int bestLab = 0x7fffffff, maxc = 0;
#pragma unroll
    for (int k = 0; k < 5; ++k) {
      int c = 0;
#pragma unroll
      for (int j = 0; j < 5; ++j) c += (lab[j] == lab[k]) ? 1 : 0;
      if (c > maxc || (c == maxc && lab[k] < bestLab)) { maxc = c; bestLab = lab[k]; }
    }
    sh_asg = bestLab;
    assigned[t] = bestLab;
    colmax[t] = sv[0];
  }
  __syncthreads();
  const int asg = sh_asg;

  // ---- phase 2: masked top-4 (positives and negatives) ----
  float vp[4]; int ip[4]; float vn[4]; int inx[4];
#pragma unroll
  for (int k = 0; k < 4; ++k) {
    vp[k] = -3.4e38f; ip[k] = 0x7fffffff;
    vn[k] = -3.4e38f; inx[k] = 0x7fffffff;
  }
  for (int q = tid; q < NSRC / 4; q += 256) {
    const F4 x = col4[q];
    const I4 l = lab4[q];
    const int s = q * 4;
    if (l.x == asg) insert_top<4>(vp, ip, x.x, s);     else insert_top<4>(vn, inx, x.x, s);
    if (l.y == asg) insert_top<4>(vp, ip, x.y, s + 1); else insert_top<4>(vn, inx, x.y, s + 1);
    if (l.z == asg) insert_top<4>(vp, ip, x.z, s + 2); else insert_top<4>(vn, inx, x.z, s + 2);
    if (l.w == asg) insert_top<4>(vp, ip, x.w, s + 3); else insert_top<4>(vn, inx, x.w, s + 3);
  }
#pragma unroll
  for (int k = 0; k < 4; ++k) { sv[tid * 4 + k] = vp[k]; si[tid * 4 + k] = ip[k]; }
  tree_merge<4>(sv, si, tid);
  if (tid == 0) {
    float nln = 0.0f;
#pragma unroll
    for (int k = 0; k < 4; ++k) if (sv[k] > -1e37f) nln += sv[k];
    sh_nln = nln;
  }
  __syncthreads();
#pragma unroll
  for (int k = 0; k < 4; ++k) { sv[tid * 4 + k] = vn[k]; si[tid * 4 + k] = inx[k]; }
  tree_merge<4>(sv, si, tid);
  if (tid == 0) {
    float nun = 0.0f;
#pragma unroll
    for (int k = 0; k < 4; ++k) if (sv[k] > -1e37f) nun += sv[k];
    scorev[t] = sh_nln / nun;
  }
}

// ---------------------------------------------------------------------------
// Kernel 4: stable top-4096 selection by rank counting (matches lax.top_k
// tie-breaking: lower index wins on equal score). Also zeroes the accumulator.
// ---------------------------------------------------------------------------
__global__ __launch_bounds__(256)
void rank_select(const float* __restrict__ scorev, int* __restrict__ sel,
                 float* __restrict__ acc) {
  __shared__ int ired[256];
  const int j = blockIdx.x;
  const int tid = threadIdx.x;
  const float my = scorev[j];
  int c = 0;
  for (int q = tid; q < NTGT; q += 256) {
    const float o = scorev[q];
    if (o > my || (o == my && q < j)) ++c;
  }
  ired[tid] = c;
  __syncthreads();
  for (int s = 128; s > 0; s >>= 1) {
    if (tid < s) ired[tid] += ired[tid + s];
    __syncthreads();
  }
  if (tid == 0) {
    sel[j] = (ired[0] < TOPSEL) ? 1 : 0;
    if (j == 0) *acc = 0.0f;
  }
}

// ---------------------------------------------------------------------------
// Kernel 5: per selected column: softmax over sources, masked ratio, log-loss.
// b128 streaming loads of both sim column and labels.
// ---------------------------------------------------------------------------
__global__ __launch_bounds__(256)
void contrastive(const float* __restrict__ simT, const int* __restrict__ labels,
                 const int* __restrict__ assigned, const float* __restrict__ colmax,
                 const int* __restrict__ sel, float* __restrict__ acc) {
  const int t = blockIdx.x;
  if (!sel[t]) return;
  const int tid = threadIdx.x;
  const F4* __restrict__ col4 = reinterpret_cast<const F4*>(simT + (size_t)t * NSRC);
  const I4* __restrict__ lab4 = reinterpret_cast<const I4*>(labels);
  const int asg = assigned[t];
  const float m = colmax[t];
  const float invtau = 1.0f / 0.07f;
  float den = 0.0f, num = 0.0f;
  for (int q = tid; q < NSRC / 4; q += 256) {
    const F4 x = col4[q];
    const I4 l = lab4[q];
    const float e0 = __expf((x.x - m) * invtau);
    const float e1 = __expf((x.y - m) * invtau);
    const float e2 = __expf((x.z - m) * invtau);
    const float e3 = __expf((x.w - m) * invtau);
    den += e0 + e1 + e2 + e3;
    if (l.x == asg) num += e0;
    if (l.y == asg) num += e1;
    if (l.z == asg) num += e2;
    if (l.w == asg) num += e3;
  }
  __shared__ float rd[256];
  __shared__ float rn[256];
  rd[tid] = den; rn[tid] = num;
  __syncthreads();
  for (int s = 128; s > 0; s >>= 1) {
    if (tid < s) { rd[tid] += rd[tid + s]; rn[tid] += rn[tid + s]; }
    __syncthreads();
  }
  if (tid == 0) atomicAdd(acc, -logf(rn[0] / rd[0] + 1e-6f));
}

__global__ void finalize(const float* __restrict__ acc, float* __restrict__ out) {
  if (threadIdx.x == 0 && blockIdx.x == 0) out[0] = acc[0] / (float)TOPSEL;
}

// ---------------------------------------------------------------------------
extern "C" void kernel_launch(void* const* d_in, const int* in_sizes, int n_in,
                              void* d_out, int out_size, void* d_ws, size_t ws_size,
                              hipStream_t stream) {
  (void)in_sizes; (void)n_in; (void)out_size; (void)ws_size;
  const float* src    = (const float*)d_in[0];
  const int*   labels = (const int*)d_in[1];
  const float* tgt    = (const float*)d_in[2];
  float* out = (float*)d_out;

  char* ws = (char*)d_ws;
  size_t off = 0;
  float* simT = (float*)(ws + off);          off += (size_t)NSRC * NTGT * sizeof(float);
  _Float16* Ahi = (_Float16*)(ws + off);     off += (size_t)NSRC * DIM * sizeof(_Float16);
  _Float16* Alo = (_Float16*)(ws + off);     off += (size_t)NSRC * DIM * sizeof(_Float16);
  _Float16* Bhi = (_Float16*)(ws + off);     off += (size_t)NTGT * DIM * sizeof(_Float16);
  _Float16* Blo = (_Float16*)(ws + off);     off += (size_t)NTGT * DIM * sizeof(_Float16);
  int*   assigned = (int*)(ws + off);        off += (size_t)NTGT * sizeof(int);
  float* colmax   = (float*)(ws + off);      off += (size_t)NTGT * sizeof(float);
  float* scorev   = (float*)(ws + off);      off += (size_t)NTGT * sizeof(float);
  int*   sel      = (int*)(ws + off);        off += (size_t)NTGT * sizeof(int);
  float* acc      = (float*)(ws + off);      off += 16;

  normalize_split<<<NSRC, 256, 0, stream>>>(src, Ahi, Alo);
  normalize_split<<<NTGT, 256, 0, stream>>>(tgt, Bhi, Blo);

  dim3 gg(NSRC / 64, NTGT / 128);
  gemm_sim<<<gg, 256, 0, stream>>>(Ahi, Alo, Bhi, Blo, simT);

  column_stats<<<NTGT, 256, 0, stream>>>(simT, labels, assigned, colmax, scorev);
  rank_select<<<NTGT, 256, 0, stream>>>(scorev, sel, acc);
  contrastive<<<NTGT, 256, 0, stream>>>(simT, labels, assigned, colmax, sel, acc);
  finalize<<<1, 32, 0, stream>>>(acc, out);
}
